// DrBC_84490596647625
// MI455X (gfx1250) — matrix operand order repaired and checked by reading, not compiled, not run on hardware
//
#include <hip/hip_runtime.h>
#include <hip/hip_bf16.h>

#define N_NODES 100000
#define N_EDGES 800000
#define EMBED   128
#define HIDDEN  32
#define GNN_BLOCKS 5

typedef __attribute__((ext_vector_type(16))) __bf16 v16bf;
typedef __attribute__((ext_vector_type(8)))  __bf16 v8bf;
typedef __attribute__((ext_vector_type(4)))  __bf16 v4bf;
typedef __attribute__((ext_vector_type(8)))  float  v8f;

// ---------------------------------------------------------------- utilities
__global__ void zero_f32_kernel(float* __restrict__ p, int n) {
    int i = blockIdx.x * blockDim.x + threadIdx.x;
    if (i < n) p[i] = 0.0f;
}

__global__ void cvt_f32_to_bf16_kernel(const float* __restrict__ src,
                                       __bf16* __restrict__ dst, int n) {
    int i = blockIdx.x * blockDim.x + threadIdx.x;
    if (i < n) dst[i] = (__bf16)src[i];
}

__global__ void deg_count_kernel(const int* __restrict__ col, float* __restrict__ deg) {
    int e = blockIdx.x * blockDim.x + threadIdx.x;
    if (e < N_EDGES) atomicAdd(&deg[col[e]], 1.0f);
}

__global__ void deg_to_dinv_kernel(float* __restrict__ deg) {
    int n = blockIdx.x * blockDim.x + threadIdx.x;
    if (n < N_NODES) deg[n] = rsqrtf(deg[n] + 1.0f);
}

__global__ void edge_norm_kernel(const int* __restrict__ row, const int* __restrict__ col,
                                 const float* __restrict__ dinv, float* __restrict__ norm) {
    int e = blockIdx.x * blockDim.x + threadIdx.x;
    if (e < N_EDGES) norm[e] = dinv[row[e]] * dinv[col[e]];
}

// h0 = relu(x @ W_embed^T + b_embed), stored bf16; also seeds layer-max z (f32).
__global__ void embed_relu_kernel(const float* __restrict__ x,
                                  const float* __restrict__ W, const float* __restrict__ b,
                                  __bf16* __restrict__ h, float* __restrict__ z) {
    int i = blockIdx.x * blockDim.x + threadIdx.x;
    if (i >= N_NODES * EMBED) return;
    int n = i >> 7, c = i & 127;
    const float* xr = x + (size_t)n * 3;
    const float* wr = W + (size_t)c * 3;
    float v = fmaf(xr[0], wr[0], fmaf(xr[1], wr[1], fmaf(xr[2], wr[2], b[c])));
    v = fmaxf(v, 0.0f);
    h[i] = (__bf16)v;
    z[i] = v;
}

// aggr[col[e]] += norm[e] * h[row[e]]   (h is bf16; aggr f32 via atomics)
__global__ void scatter_msg_kernel(const int* __restrict__ row, const int* __restrict__ col,
                                   const float* __restrict__ norm,
                                   const __bf16* __restrict__ h, float* __restrict__ aggr) {
    long long t = (long long)blockIdx.x * blockDim.x + threadIdx.x;
    if (t >= (long long)N_EDGES * 32) return;
    int e  = (int)(t >> 5);
    int c4 = ((int)t & 31) << 2;
    float nm = norm[e];
    const v4bf hv = *(const v4bf*)(h + (size_t)row[e] * EMBED + c4);
    float* dst = aggr + (size_t)col[e] * EMBED + c4;
    atomicAdd(dst + 0, nm * (float)hv[0]);
    atomicAdd(dst + 1, nm * (float)hv[1]);
    atomicAdd(dst + 2, nm * (float)hv[2]);
    atomicAdd(dst + 3, nm * (float)hv[3]);
}

__device__ __forceinline__ float sigmoidf_(float v) { return 1.0f / (1.0f + __expf(-v)); }

__device__ __forceinline__ v16bf cat8(v8bf lo, v8bf hi) {
    return __builtin_shufflevector(lo, hi, 0, 1, 2, 3, 4, 5, 6, 7,
                                           8, 9, 10, 11, 12, 13, 14, 15);
}

// ------------------------------------------------ fused GRU block via WMMA
// Wave tile: 32 nodes (2 M sub-tiles) x 16 channels.
// gi = hprev @ W_ih^T + b_ih ; gh = aggr @ W_hh^T + b_hh
// r = sig(i_r+h_r); z = sig(i_z+h_z); n = tanh(i_n + r*h_n)
// h_new = (1-z)*n + z*aggr (bf16 out) ; zmax = max(zmax, h_new) (f32)
__global__ __launch_bounds__(256) void gru_wmma_kernel(
    const __bf16* __restrict__ hprev, const float* __restrict__ aggr,
    const __bf16* __restrict__ Wih,  const __bf16* __restrict__ Whh,
    const float* __restrict__ b_ih,  const float* __restrict__ b_hh,
    __bf16* __restrict__ hnext, float* __restrict__ zmax) {

    const int lane   = threadIdx.x & 31;
    const int wave   = threadIdx.x >> 5;       // 8 waves -> 8 channel tiles
    const int m_base = blockIdx.x * 32;        // 32-node tile
    const int c_base = wave * 16;              // channel tile within EMBED
    const bool hi    = lane >= 16;
    const int  l15   = lane & 15;

    v8f zerov = {0.f, 0.f, 0.f, 0.f, 0.f, 0.f, 0.f, 0.f};
    v8f acc[2][6];
#pragma unroll
    for (int t = 0; t < 2; ++t)
#pragma unroll
        for (int g = 0; g < 6; ++g) acc[t][g] = zerov;

    // A layout (16-bit 16x32): lane<16 -> row=l15, K runs {0..7,16..23}; hi lanes +8.
    const int   kAoff = hi ? 8 : 0;
    const __bf16* hrow0 = hprev + (size_t)(m_base + l15) * EMBED + kAoff;
    const __bf16* hrow1 = hrow0 + (size_t)16 * EMBED;
    const float*  grow0 = aggr  + (size_t)(m_base + l15) * EMBED + kAoff;
    const float*  grow1 = grow0 + (size_t)16 * EMBED;
    // B layout (16-bit 32x16): lane -> col N=l15, K runs {0..7,16..23}; hi lanes +8.
    const int nW    = c_base + l15;
    const int kBoff = hi ? 8 : 0;

#pragma unroll
    for (int kc = 0; kc < 4; ++kc) {
        const int k0 = kc * 32;
        // A (hprev): direct bf16 16B loads
        v16bf aH0 = cat8(*(const v8bf*)(hrow0 + k0), *(const v8bf*)(hrow0 + k0 + 16));
        v16bf aH1 = cat8(*(const v8bf*)(hrow1 + k0), *(const v8bf*)(hrow1 + k0 + 16));
        // A (aggr): f32 -> bf16 in-register
        v16bf aG0, aG1;
#pragma unroll
        for (int i = 0; i < 8; ++i) {
            aG0[i]     = (__bf16)grow0[k0 + i];
            aG0[i + 8] = (__bf16)grow0[k0 + 16 + i];
            aG1[i]     = (__bf16)grow1[k0 + i];
            aG1[i + 8] = (__bf16)grow1[k0 + 16 + i];
        }
#pragma unroll
        for (int g = 0; g < 3; ++g) {
            const __bf16* wi = Wih + (size_t)(g * EMBED + nW) * EMBED + k0 + kBoff;
            const __bf16* wh = Whh + (size_t)(g * EMBED + nW) * EMBED + k0 + kBoff;
            v16bf bi = cat8(*(const v8bf*)(wi), *(const v8bf*)(wi + 16));
            v16bf bh = cat8(*(const v8bf*)(wh), *(const v8bf*)(wh + 16));
            acc[0][g]     = __builtin_amdgcn_wmma_f32_16x16x32_bf16(
                                false, aH0, false, bi, (short)0, acc[0][g], false, false);
            acc[1][g]     = __builtin_amdgcn_wmma_f32_16x16x32_bf16(
                                false, aH1, false, bi, (short)0, acc[1][g], false, false);
            acc[0][3 + g] = __builtin_amdgcn_wmma_f32_16x16x32_bf16(
                                false, aG0, false, bh, (short)0, acc[0][3 + g], false, false);
            acc[1][3 + g] = __builtin_amdgcn_wmma_f32_16x16x32_bf16(
                                false, aG1, false, bh, (short)0, acc[1][3 + g], false, false);
        }
    }

    // C/D layout: VGPR r -> (M = base + r + 8*hi, N = c_base + l15)
    const int nC = c_base + l15;
    const float bir = b_ih[nC], biz = b_ih[EMBED + nC], bin = b_ih[2 * EMBED + nC];
    const float bhr = b_hh[nC], bhz = b_hh[EMBED + nC], bhn = b_hh[2 * EMBED + nC];
#pragma unroll
    for (int t = 0; t < 2; ++t) {
#pragma unroll
        for (int r = 0; r < 8; ++r) {
            const int mC = m_base + t * 16 + r + (hi ? 8 : 0);
            const size_t idx = (size_t)mC * EMBED + nC;
            float rr = sigmoidf_(acc[t][0][r] + bir + acc[t][3][r] + bhr);
            float zz = sigmoidf_(acc[t][1][r] + biz + acc[t][4][r] + bhz);
            float nn = tanhf(acc[t][2][r] + bin + rr * (acc[t][5][r] + bhn));
            float ag = aggr[idx];
            float hv = (1.0f - zz) * nn + zz * ag;
            hnext[idx] = (__bf16)hv;
            zmax[idx]  = fmaxf(zmax[idx], hv);
        }
    }
}

// ------------------------------------------------ decoder: one wave32 / node
__global__ void decoder_kernel(const float* __restrict__ z,
                               const float* __restrict__ W_dec, const float* __restrict__ b_dec,
                               const float* __restrict__ W_out, const float* __restrict__ b_out,
                               float* __restrict__ out) {
    long long t = (long long)blockIdx.x * blockDim.x + threadIdx.x;
    int node = (int)(t >> 5);
    int j    = (int)t & 31;          // hidden index, wave32 lane
    if (node >= N_NODES) return;
    const float* zr = z + (size_t)node * EMBED;
    const float* wr = W_dec + (size_t)j * EMBED;
    float acc = b_dec[j];
#pragma unroll 8
    for (int k = 0; k < EMBED; ++k) acc = fmaf(zr[k], wr[k], acc);
    float d = fmaxf(acc, 0.0f) * W_out[j];
#pragma unroll
    for (int off = 16; off > 0; off >>= 1) d += __shfl_xor(d, off, 32);
    if (j == 0) out[node] = d + b_out[0];
}

// ---------------------------------------------------------------- launcher
static inline char* align256(char* p) {
    return (char*)(((uintptr_t)p + 255) & ~(uintptr_t)255);
}

extern "C" void kernel_launch(void* const* d_in, const int* in_sizes, int n_in,
                              void* d_out, int out_size, void* d_ws, size_t ws_size,
                              hipStream_t stream) {
    (void)in_sizes; (void)n_in; (void)out_size; (void)ws_size;
    const float* x       = (const float*)d_in[0];
    const int*   edge    = (const int*)d_in[1];
    const float* W_embed = (const float*)d_in[2];
    const float* b_embed = (const float*)d_in[3];
    const float* W_ih    = (const float*)d_in[4];
    const float* W_hh    = (const float*)d_in[5];
    const float* b_ih    = (const float*)d_in[6];
    const float* b_hh    = (const float*)d_in[7];
    const float* W_dec   = (const float*)d_in[8];
    const float* b_dec   = (const float*)d_in[9];
    const float* W_out   = (const float*)d_in[10];
    const float* b_out   = (const float*)d_in[11];
    const int* row = edge;
    const int* col = edge + N_EDGES;

    const size_t NH = (size_t)N_NODES * EMBED;
    const int    NW = 3 * EMBED * EMBED;   // elements per GRU weight matrix

    char* ws = (char*)d_ws;
    float* norm  = (float*)ws;  ws = align256(ws + sizeof(float) * (size_t)N_EDGES);
    float* dinv  = (float*)ws;  ws = align256(ws + sizeof(float) * (size_t)N_NODES);
    float* zbuf  = (float*)ws;  ws = align256(ws + sizeof(float) * NH);
    float* aggr  = (float*)ws;  ws = align256(ws + sizeof(float) * NH);
    __bf16* hA   = (__bf16*)ws; ws = align256(ws + sizeof(__bf16) * NH);
    __bf16* hB   = (__bf16*)ws; ws = align256(ws + sizeof(__bf16) * NH);
    __bf16* WihB = (__bf16*)ws; ws = align256(ws + sizeof(__bf16) * (size_t)NW);
    __bf16* WhhB = (__bf16*)ws; ws = align256(ws + sizeof(__bf16) * (size_t)NW);

    const int NE_BLK  = (N_EDGES + 255) / 256;
    const int NN_BLK  = (N_NODES + 255) / 256;
    const int NH_BLK  = (int)((NH + 255) / 256);
    const int NW_BLK  = (NW + 255) / 256;
    const int SC_BLK  = (int)(((long long)N_EDGES * 32 + 255) / 256);
    const int DEC_BLK = (int)(((long long)N_NODES * 32 + 255) / 256);

    // weights -> bf16 (once per launch)
    cvt_f32_to_bf16_kernel<<<NW_BLK, 256, 0, stream>>>(W_ih, WihB, NW);
    cvt_f32_to_bf16_kernel<<<NW_BLK, 256, 0, stream>>>(W_hh, WhhB, NW);

    // degree -> dinv -> per-edge norm
    zero_f32_kernel<<<NN_BLK, 256, 0, stream>>>(dinv, N_NODES);
    deg_count_kernel<<<NE_BLK, 256, 0, stream>>>(col, dinv);
    deg_to_dinv_kernel<<<NN_BLK, 256, 0, stream>>>(dinv);
    edge_norm_kernel<<<NE_BLK, 256, 0, stream>>>(row, col, dinv, norm);

    // embed + seed layer max
    embed_relu_kernel<<<NH_BLK, 256, 0, stream>>>(x, W_embed, b_embed, hA, zbuf);

    __bf16* hp = hA;
    __bf16* hn = hB;
    for (int b = 0; b < GNN_BLOCKS; ++b) {
        zero_f32_kernel<<<NH_BLK, 256, 0, stream>>>(aggr, (int)NH);
        scatter_msg_kernel<<<SC_BLK, 256, 0, stream>>>(row, col, norm, hp, aggr);
        gru_wmma_kernel<<<N_NODES / 32, 256, 0, stream>>>(hp, aggr, WihB, WhhB,
                                                          b_ih, b_hh, hn, zbuf);
        __bf16* t = hp; hp = hn; hn = t;
    }

    decoder_kernel<<<DEC_BLK, 256, 0, stream>>>(zbuf, W_dec, b_dec, W_out, b_out,
                                                (float*)d_out);
}
